// WanAttentionBlock_2302102471426
// MI455X (gfx1250) — matrix operand order repaired and checked by reading, not compiled, not run on hardware
//
#include <hip/hip_runtime.h>
#include <hip/hip_bf16.h>
#include <math.h>

// ---------------------------------------------------------------------------
// Problem constants (B=1)
// ---------------------------------------------------------------------------
constexpr int S_LEN = 3072;
constexpr int DMODEL = 1536;
constexpr int NHEAD = 12;
constexpr int HDIM = 128;
constexpr int FDIM = 8960;

typedef __attribute__((ext_vector_type(16))) _Float16 v16h;
typedef __attribute__((ext_vector_type(8)))  float    v8f;
typedef __attribute__((ext_vector_type(8)))  unsigned short ush8;

__device__ inline v8f vzero8() {
    v8f z;
#pragma unroll
    for (int i = 0; i < 8; ++i) z[i] = 0.f;
    return z;
}

// ---------------------------------------------------------------------------
// CDNA5 async copy: global -> LDS, 16B per lane, tracked by ASYNCcnt.
// VDST = LDS byte offset (VGPR), VADDR = 64-bit global address, SADDR = off.
// Generic LDS pointers carry the LDS offset in their low 32 bits (ISA §10.2).
// ---------------------------------------------------------------------------
__device__ inline unsigned lds_off(const void* p) {
    return (unsigned)(uintptr_t)p;
}

__device__ inline void async_ld16(unsigned lds_addr, const void* gptr) {
    asm volatile("global_load_async_to_lds_b128 %0, %1, off"
                 :: "v"(lds_addr), "v"((unsigned long long)(uintptr_t)gptr)
                 : "memory");
}

__device__ inline void wait_async0() {
    asm volatile("s_wait_asynccnt 0x0" ::: "memory");
}

// ---------------------------------------------------------------------------
// WMMA fragment loaders (16-bit operands; this code's contiguous k order, same in A and B)
// A: 16x32 (MxK). lane L: m = L%16, half h = L/16.
//    slot t -> K = h*16 + t (16 contiguous halfs)
// B: 32x16 (KxN) stored as [n][k] rows in LDS. lane L: n = L%16, h = L/16.
//    slot t -> K = h*16 + t (16 contiguous halfs)
// C/D: v8f. element e: row = e + 8*h, col = L%16.
// ---------------------------------------------------------------------------
__device__ inline v16h load_frag_a(const _Float16* lds, int row0, int k0, int ld) {
    const int lane = threadIdx.x & 31;
    const int m = row0 + (lane & 15);
    const int h = lane >> 4;
    const _Float16* p = lds + m * ld + k0;
    v16h a;
#pragma unroll
    for (int t = 0; t < 8; ++t) a[t] = p[h * 16 + t];
#pragma unroll
    for (int t = 0; t < 8; ++t) a[8 + t] = p[h * 16 + 8 + t];
    return a;
}

__device__ inline v16h load_frag_b(const _Float16* lds, int n0, int k0, int ld) {
    const int lane = threadIdx.x & 31;
    const int n = n0 + (lane & 15);
    const int h = lane >> 4;
    const _Float16* p = lds + n * ld + k0 + h * 16;
    v16h b;
#pragma unroll
    for (int t = 0; t < 16; ++t) b[t] = p[t];
    return b;
}

// ---------------------------------------------------------------------------
// fp32 -> f16 conversion (grid-stride)
// ---------------------------------------------------------------------------
__global__ void k_cvt_f16(const float* __restrict__ src, _Float16* __restrict__ dst, size_t n) {
    size_t i = (size_t)blockIdx.x * blockDim.x + threadIdx.x;
    size_t stride = (size_t)gridDim.x * blockDim.x;
    for (; i < n; i += stride) dst[i] = (_Float16)src[i];
}

// ---------------------------------------------------------------------------
// Fused LayerNorm + adaLN modulation: out = shift + ln(x)*(1+scale), f16 out.
// ---------------------------------------------------------------------------
__global__ __launch_bounds__(256)
void k_ln_mod(const float* __restrict__ x, const float* __restrict__ e,
              const float* __restrict__ mod, int shiftRow, int scaleRow,
              _Float16* __restrict__ out) {
    const int s = blockIdx.x;
    const float* xr = x + (size_t)s * DMODEL;
    __shared__ float rs[256], rq[256];
    __shared__ float mu_s, rstd_s;
    float sum = 0.f, ssq = 0.f;
    for (int d = threadIdx.x; d < DMODEL; d += 256) {
        float v = xr[d];
        sum += v; ssq += v * v;
    }
    rs[threadIdx.x] = sum; rq[threadIdx.x] = ssq;
    __syncthreads();
    for (int off = 128; off > 0; off >>= 1) {
        if (threadIdx.x < off) {
            rs[threadIdx.x] += rs[threadIdx.x + off];
            rq[threadIdx.x] += rq[threadIdx.x + off];
        }
        __syncthreads();
    }
    if (threadIdx.x == 0) {
        float mu = rs[0] / DMODEL;
        float var = rq[0] / DMODEL - mu * mu;
        mu_s = mu;
        rstd_s = rsqrtf(var + 1e-6f);
    }
    __syncthreads();
    const float mu = mu_s, rstd = rstd_s;
    for (int d = threadIdx.x; d < DMODEL; d += 256) {
        float sh = mod[(size_t)shiftRow * DMODEL + d] + e[(size_t)shiftRow * DMODEL + d];
        float sc = mod[(size_t)scaleRow * DMODEL + d] + e[(size_t)scaleRow * DMODEL + d];
        float v = (xr[d] - mu) * rstd;
        out[(size_t)s * DMODEL + d] = (_Float16)(sh + v * (1.f + sc));
    }
}

// ---------------------------------------------------------------------------
// Tiled WMMA GEMM: C[M x Nout] = A_f16[M x K] @ W_f16[Nout x K]^T + bias
// Macro tile 128x128, K-step 32, double-buffered LDS filled by
// global_load_async_to_lds_b128. 8 waves as 4x2; wave owns a 32x64 tile
// (2 A frags x 4 B frags -> 8 WMMAs per K-step).
// EPI: 0 = f32 out (bias), 1 = f16 out (bias + tanh-GELU), 2 = f16 out (bias).
// ---------------------------------------------------------------------------
template <int EPI, typename OT>
__global__ __launch_bounds__(256)
void k_gemm(const _Float16* __restrict__ A, const _Float16* __restrict__ W,
            const float* __restrict__ bias, OT* __restrict__ C,
            int M, int Nout, int K) {
    __shared__ _Float16 As[2][128 * 40];
    __shared__ _Float16 Ws[2][128 * 40];
    const int tid = threadIdx.x;
    const int wave = tid >> 5;
    const int lane = tid & 31;
    const int rg = wave >> 1;      // 0..3: 32-row group
    const int cg = wave & 1;       // 0..1: 64-col group
    const int m0 = blockIdx.y * 128;
    const int n0 = blockIdx.x * 128;
    (void)M;

    v8f acc[2][4];
#pragma unroll
    for (int i = 0; i < 2; ++i)
#pragma unroll
        for (int j = 0; j < 4; ++j) acc[i][j] = vzero8();

    // per-thread 16B chunk coords: 128x32 halfs = 512 chunks of 8; thread does 2
    const int r0 = tid >> 2, k80 = (tid & 3) * 8;
    const int r1 = (tid + 256) >> 2, k81 = ((tid + 256) & 3) * 8;

    auto issue = [&](int buf, int kk) {
        async_ld16(lds_off(&As[buf][r0 * 40 + k80]), A + (size_t)(m0 + r0) * K + kk + k80);
        async_ld16(lds_off(&As[buf][r1 * 40 + k81]), A + (size_t)(m0 + r1) * K + kk + k81);
        async_ld16(lds_off(&Ws[buf][r0 * 40 + k80]), W + (size_t)(n0 + r0) * K + kk + k80);
        async_ld16(lds_off(&Ws[buf][r1 * 40 + k81]), W + (size_t)(n0 + r1) * K + kk + k81);
    };

    issue(0, 0);
    int buf = 0;
    for (int kk = 0; kk < K; kk += 32) {
        wait_async0();          // this thread's current-buffer loads landed in LDS
        __syncthreads();        // => everyone's landed; prev math fully done
        if (kk + 32 < K) issue(buf ^ 1, kk + 32);   // overlap next loads with math

        v16h af[2], bf[4];
#pragma unroll
        for (int mt = 0; mt < 2; ++mt)
            af[mt] = load_frag_a(As[buf], rg * 32 + mt * 16, 0, 40);
#pragma unroll
        for (int nt = 0; nt < 4; ++nt)
            bf[nt] = load_frag_b(Ws[buf], cg * 64 + nt * 16, 0, 40);
#pragma unroll
        for (int mt = 0; mt < 2; ++mt)
#pragma unroll
            for (int nt = 0; nt < 4; ++nt)
                acc[mt][nt] = __builtin_amdgcn_wmma_f32_16x16x32_f16(
                    false, af[mt], false, bf[nt], (short)0, acc[mt][nt], false, false);
        buf ^= 1;
    }

    const int h = lane >> 4, nlo = lane & 15;
#pragma unroll
    for (int mt = 0; mt < 2; ++mt) {
#pragma unroll
        for (int nt = 0; nt < 4; ++nt) {
            int col = n0 + cg * 64 + nt * 16 + nlo;
            float bv = bias[col];
#pragma unroll
            for (int e2 = 0; e2 < 8; ++e2) {
                int row = m0 + rg * 32 + mt * 16 + e2 + 8 * h;
                float v = acc[mt][nt][e2] + bv;
                if (EPI == 1) {
                    float t = 0.7978845608f * (v + 0.044715f * v * v * v);
                    v = 0.5f * v * (1.f + tanhf(t));
                }
                C[(size_t)row * Nout + col] = (OT)v;
            }
        }
    }
}

// ---------------------------------------------------------------------------
// Fused RMSNorm (over D) + RoPE (per head of HD), f32 in -> f16 out.
// ---------------------------------------------------------------------------
__global__ __launch_bounds__(256)
void k_rms_rope(const float* __restrict__ q, const float* __restrict__ nw,
                const float* __restrict__ freqs, _Float16* __restrict__ out) {
    const int s = blockIdx.x;
    const float* qr = q + (size_t)s * DMODEL;
    __shared__ float rq[256];
    __shared__ float rr_s;
    float ssq = 0.f;
    for (int d = threadIdx.x; d < DMODEL; d += 256) {
        float v = qr[d];
        ssq += v * v;
    }
    rq[threadIdx.x] = ssq;
    __syncthreads();
    for (int off = 128; off > 0; off >>= 1) {
        if (threadIdx.x < off) rq[threadIdx.x] += rq[threadIdx.x + off];
        __syncthreads();
    }
    if (threadIdx.x == 0) rr_s = rsqrtf(rq[0] / DMODEL + 1e-6f);
    __syncthreads();
    const float rr = rr_s;
    const float* fs = freqs + (size_t)s * HDIM;  // HD/2 pairs * {cos,sin}
    for (int pp = threadIdx.x; pp < DMODEL / 2; pp += 256) {
        int d0 = 2 * pp;
        int p = pp & (HDIM / 2 - 1);
        float c = fs[p * 2 + 0];
        float sn = fs[p * 2 + 1];
        float x0 = qr[d0] * rr * nw[d0];
        float x1 = qr[d0 + 1] * rr * nw[d0 + 1];
        out[(size_t)s * DMODEL + d0]     = (_Float16)(x0 * c - x1 * sn);
        out[(size_t)s * DMODEL + d0 + 1] = (_Float16)(x0 * sn + x1 * c);
    }
}

// ---------------------------------------------------------------------------
// Flash attention with WMMA. Block = (64 queries, 1 head), 256 threads (8 waves).
// Q and K tiles staged via global_load_async_to_lds_b128 (ASYNCcnt path);
// V is transposed lane-wise into LDS (sync) with a prefetch hint for the next
// tile. Online softmax state (m, l, alpha) lives in LDS.
// ---------------------------------------------------------------------------
__global__ __launch_bounds__(256)
void k_attn(const _Float16* __restrict__ Qh, const _Float16* __restrict__ Kh,
            const _Float16* __restrict__ Vh, const int* __restrict__ seq_lens,
            _Float16* __restrict__ Yh) {
    __shared__ _Float16 Qs[64 * 136];
    __shared__ _Float16 KV[128 * 72];   // K tile [key][136] or V^T [hd][72]
    __shared__ float    Sbuf[64 * 66];
    __shared__ _Float16 Pbuf[64 * 72];
    __shared__ float m_s[64], l_s[64], a_s[64];

    const int tid = threadIdx.x;
    const int wave = tid >> 5;
    const int lane = tid & 31;
    const int rg = wave >> 1;
    const int cg = wave & 1;
    const int q0 = blockIdx.x * 64;
    const int head = blockIdx.y;
    const int seqlen = seq_lens[0];
    const int hlane = lane >> 4, nlo = lane & 15;

    // Q tile 64x128 via async copy (4 x 16B per thread)
#pragma unroll
    for (int i = 0; i < 4; ++i) {
        int c = tid + i * 256;
        int r = c >> 4, k8 = (c & 15) * 8;
        async_ld16(lds_off(&Qs[r * 136 + k8]),
                   Qh + (size_t)(q0 + r) * DMODEL + head * HDIM + k8);
    }
    if (tid < 64) { m_s[tid] = -1e30f; l_s[tid] = 0.f; }

    v8f oacc[4];
#pragma unroll
    for (int i = 0; i < 4; ++i) oacc[i] = vzero8();

    int nkt = (seqlen + 63) >> 6;
    if (nkt > S_LEN / 64) nkt = S_LEN / 64;

    for (int kt = 0; kt < nkt; ++kt) {
        // K tile 64x128 via async copy; safe: end-of-iteration barrier below
        // guarantees all waves are done reading KV (as V^T) from last round.
#pragma unroll
        for (int i = 0; i < 4; ++i) {
            int c = tid + i * 256;
            int r = c >> 4, k8 = (c & 15) * 8;
            async_ld16(lds_off(&KV[r * 136 + k8]),
                       Kh + (size_t)(kt * 64 + r) * DMODEL + head * HDIM + k8);
        }
        if (kt + 1 < nkt)  // prefetch hint for next V tile (stays sync-loaded)
            __builtin_prefetch(Vh + (size_t)((kt + 1) * 64 + (tid >> 2)) * DMODEL + head * HDIM, 0, 0);
        wait_async0();
        __syncthreads();

        v8f sacc[2];
        sacc[0] = vzero8(); sacc[1] = vzero8();
#pragma unroll
        for (int k0 = 0; k0 < 4; ++k0) {
            v16h afrag = load_frag_a(Qs, rg * 16, k0 * 32, 136);
#pragma unroll
            for (int n = 0; n < 2; ++n) {
                v16h bfrag = load_frag_b(KV, cg * 32 + n * 16, k0 * 32, 136);
                sacc[n] = __builtin_amdgcn_wmma_f32_16x16x32_f16(
                    false, afrag, false, bfrag, (short)0, sacc[n], false, false);
            }
        }
        const float scale = 0.08838834765f;  // 1/sqrt(128)
#pragma unroll
        for (int n = 0; n < 2; ++n) {
#pragma unroll
            for (int e2 = 0; e2 < 8; ++e2) {
                int row = rg * 16 + e2 + 8 * hlane;
                int colL = cg * 32 + n * 16 + nlo;
                float v = sacc[n][e2] * scale;
                if (kt * 64 + colL >= seqlen) v = -1e9f;
                Sbuf[row * 66 + colL] = v;
            }
        }
        __syncthreads();

        if (tid < 64) {
            int row = tid;
            float mo = m_s[row], mn = mo;
            for (int c = 0; c < 64; ++c) mn = fmaxf(mn, Sbuf[row * 66 + c]);
            float alpha = __expf(mo - mn);
            float ls = 0.f;
            for (int c = 0; c < 64; ++c) {
                float p = __expf(Sbuf[row * 66 + c] - mn);
                ls += p;
                Pbuf[row * 72 + c] = (_Float16)p;
            }
            l_s[row] = l_s[row] * alpha + ls;
            m_s[row] = mn;
            a_s[row] = alpha;
        }
        __syncthreads();

        // V tile 64x128 -> transposed into KV [hd][72]
        for (int c = tid; c < 64 * 16; c += 256) {
            int r = c >> 4, k8 = (c & 15) * 8;
            ush8 v = *(const ush8*)(Vh + (size_t)(kt * 64 + r) * DMODEL + head * HDIM + k8);
            const _Float16* pv = (const _Float16*)&v;
#pragma unroll
            for (int j = 0; j < 8; ++j) KV[(k8 + j) * 72 + r] = pv[j];
        }
        __syncthreads();

#pragma unroll
        for (int n = 0; n < 4; ++n) {
#pragma unroll
            for (int e2 = 0; e2 < 8; ++e2) {
                oacc[n][e2] *= a_s[rg * 16 + e2 + 8 * hlane];
            }
        }
#pragma unroll
        for (int k0 = 0; k0 < 2; ++k0) {
            v16h afrag = load_frag_a(Pbuf, rg * 16, k0 * 32, 72);
#pragma unroll
            for (int n = 0; n < 4; ++n) {
                v16h bfrag = load_frag_b(KV, cg * 64 + n * 16, k0 * 32, 72);
                oacc[n] = __builtin_amdgcn_wmma_f32_16x16x32_f16(
                    false, afrag, false, bfrag, (short)0, oacc[n], false, false);
            }
        }
        __syncthreads();
    }

#pragma unroll
    for (int n = 0; n < 4; ++n) {
#pragma unroll
        for (int e2 = 0; e2 < 8; ++e2) {
            int row = rg * 16 + e2 + 8 * hlane;
            int col = cg * 64 + n * 16 + nlo;
            float inv = 1.f / l_s[row];
            Yh[(size_t)(q0 + row) * DMODEL + head * HDIM + col] = (_Float16)(oacc[n][e2] * inv);
        }
    }
}

// ---------------------------------------------------------------------------
// Residual + gate: dst = x + y * (mod[g] + e[g])
// ---------------------------------------------------------------------------
__global__ void k_res_gate(const float* __restrict__ x, const float* __restrict__ y,
                           const float* __restrict__ e, const float* __restrict__ mod,
                           int gateRow, float* __restrict__ dst, size_t n) {
    size_t i = (size_t)blockIdx.x * blockDim.x + threadIdx.x;
    size_t stride = (size_t)gridDim.x * blockDim.x;
    for (; i < n; i += stride) {
        int d = (int)(i % DMODEL);
        float g = mod[(size_t)gateRow * DMODEL + d] + e[(size_t)gateRow * DMODEL + d];
        dst[i] = x[i] + y[i] * g;
    }
}

// ---------------------------------------------------------------------------
extern "C" void kernel_launch(void* const* d_in, const int* in_sizes, int n_in,
                              void* d_out, int out_size, void* d_ws, size_t ws_size,
                              hipStream_t stream) {
    const float* x    = (const float*)d_in[0];
    const float* e    = (const float*)d_in[1];
    const int*   slen = (const int*)d_in[2];
    const float* fr   = (const float*)d_in[3];
    const float* mod  = (const float*)d_in[4];
    const float* qw   = (const float*)d_in[5];
    const float* qb   = (const float*)d_in[6];
    const float* kw   = (const float*)d_in[7];
    const float* kb   = (const float*)d_in[8];
    const float* vw   = (const float*)d_in[9];
    const float* vb   = (const float*)d_in[10];
    const float* ow   = (const float*)d_in[11];
    const float* ob   = (const float*)d_in[12];
    const float* nqw  = (const float*)d_in[13];
    const float* nkw  = (const float*)d_in[14];
    const float* f1w  = (const float*)d_in[15];
    const float* f1b  = (const float*)d_in[16];
    const float* f2w  = (const float*)d_in[17];
    const float* f2b  = (const float*)d_in[18];
    float* out = (float*)d_out;

    char* wsb = (char*)d_ws;
    size_t off = 0;
    auto take = [&](size_t bytes) -> char* {
        char* p = wsb + off;
        off += (bytes + 255) & ~(size_t)255;
        return p;
    };
    const size_t SD = (size_t)S_LEN * DMODEL;
    const size_t DD = (size_t)DMODEL * DMODEL;
    const size_t FD = (size_t)FDIM * DMODEL;
    const size_t SF = (size_t)S_LEN * FDIM;

    _Float16* inp_h = (_Float16*)take(SD * 2);
    _Float16* wq_h  = (_Float16*)take(DD * 2);
    _Float16* wk_h  = (_Float16*)take(DD * 2);
    _Float16* wv_h  = (_Float16*)take(DD * 2);
    _Float16* wo_h  = (_Float16*)take(DD * 2);
    _Float16* wf1_h = (_Float16*)take(FD * 2);
    _Float16* wf2_h = (_Float16*)take(FD * 2);
    float*    tmp32 = (float*)take(SD * 4);
    _Float16* qh    = (_Float16*)take(SD * 2);
    _Float16* kh    = (_Float16*)take(SD * 2);
    _Float16* vh    = (_Float16*)take(SD * 2);
    _Float16* yh    = (_Float16*)take(SD * 2);
    float*    xnew  = (float*)take(SD * 4);
    _Float16* hbuf  = (_Float16*)take(SF * 2);
    (void)ws_size; (void)in_sizes; (void)n_in; (void)out_size;

    const dim3 b256(256);

    k_cvt_f16<<<4096, b256, 0, stream>>>(qw,  wq_h,  DD);
    k_cvt_f16<<<4096, b256, 0, stream>>>(kw,  wk_h,  DD);
    k_cvt_f16<<<4096, b256, 0, stream>>>(vw,  wv_h,  DD);
    k_cvt_f16<<<4096, b256, 0, stream>>>(ow,  wo_h,  DD);
    k_cvt_f16<<<4096, b256, 0, stream>>>(f1w, wf1_h, FD);
    k_cvt_f16<<<4096, b256, 0, stream>>>(f2w, wf2_h, FD);

    k_ln_mod<<<S_LEN, b256, 0, stream>>>(x, e, mod, 0, 1, inp_h);

    const dim3 gProj(DMODEL / 128, S_LEN / 128);
    k_gemm<0, float><<<gProj, b256, 0, stream>>>(inp_h, wq_h, qb, tmp32, S_LEN, DMODEL, DMODEL);
    k_rms_rope<<<S_LEN, b256, 0, stream>>>(tmp32, nqw, fr, qh);
    k_gemm<0, float><<<gProj, b256, 0, stream>>>(inp_h, wk_h, kb, tmp32, S_LEN, DMODEL, DMODEL);
    k_rms_rope<<<S_LEN, b256, 0, stream>>>(tmp32, nkw, fr, kh);
    k_gemm<2, _Float16><<<gProj, b256, 0, stream>>>(inp_h, wv_h, vb, vh, S_LEN, DMODEL, DMODEL);

    k_attn<<<dim3(S_LEN / 64, NHEAD), b256, 0, stream>>>(qh, kh, vh, slen, yh);

    k_gemm<0, float><<<gProj, b256, 0, stream>>>(yh, wo_h, ob, tmp32, S_LEN, DMODEL, DMODEL);
    k_res_gate<<<4096, b256, 0, stream>>>(x, tmp32, e, mod, 2, xnew, SD);

    k_ln_mod<<<S_LEN, b256, 0, stream>>>(xnew, e, mod, 3, 4, inp_h);

    k_gemm<1, _Float16><<<dim3(FDIM / 128, S_LEN / 128), b256, 0, stream>>>(
        inp_h, wf1_h, f1b, hbuf, S_LEN, FDIM, DMODEL);
    k_gemm<0, float><<<gProj, b256, 0, stream>>>(hbuf, wf2_h, f2b, tmp32, S_LEN, DMODEL, FDIM);

    k_res_gate<<<4096, b256, 0, stream>>>(xnew, tmp32, e, mod, 5, out, SD);
}
